// MockFusedMoE_21199958573479
// MI455X (gfx1250) — matrix operand order repaired and checked by reading, not compiled, not run on hardware
//
#include <hip/hip_runtime.h>
#include <hip/hip_bf16.h>

#define T_TOK 8192
#define DMODEL 1024
#define NEXP 16
#define IDIM 2048
#define TOPK 4

#define TM 32
#define XS_STRIDE (DMODEL + 8)   // bf16 elems; +16B pad to spread LDS banks
#define HS_STRIDE (IDIM + 8)
#define NWAVE 8

typedef __attribute__((ext_vector_type(16))) __bf16 v16bf;
typedef __attribute__((ext_vector_type(8)))  float  v8f;

union Frag {
  v16bf  v;
  __bf16 h[16];
  unsigned int u[8];
};

__device__ __forceinline__ v8f zero8() {
  v8f z = {0.f, 0.f, 0.f, 0.f, 0.f, 0.f, 0.f, 0.f};
  return z;
}

__device__ __forceinline__ v8f wmma_bf16(v16bf a, v16bf b, v8f c) {
  return __builtin_amdgcn_wmma_f32_16x16x32_bf16(false, a, false, b, (short)0, c,
                                                 false, false);
}

// Fast SiLU: g * sigmoid(g) using v_exp_f32 + v_rcp_f32 (result is truncated
// to bf16 anyway, so the ~1-ulp rcp approximation is lossless here).
__device__ __forceinline__ float silu_mul(float g, float u) {
  float s = __builtin_amdgcn_rcpf(1.f + __expf(-g));
  return g * s * u;
}

// A-matrix 16-bit 16x32 layout: lane L holds row M=L&15;
// half hi=L>>4 selects K = {hi*8..hi*8+7} and {16+hi*8..16+hi*8+7}.
__device__ __forceinline__ v16bf frag_a_lds(const __bf16* row, int k0, int hi) {
  Frag f;
  const uint4* p0 = (const uint4*)(row + k0 + hi * 8);
  const uint4* p1 = (const uint4*)(row + k0 + 16 + hi * 8);
  uint4 q0 = *p0;
  uint4 q1 = *p1;
  f.u[0] = q0.x; f.u[1] = q0.y; f.u[2] = q0.z; f.u[3] = q0.w;
  f.u[4] = q1.x; f.u[5] = q1.y; f.u[6] = q1.z; f.u[7] = q1.w;
  return f.v;
}

// B-matrix 16-bit 32x16 layout: lane L holds column N=L&15,
// K = {hi*8..hi*8+7} and {16+hi*8..16+hi*8+7}. Source rows are f32, convert to bf16.
__device__ __forceinline__ v16bf frag_b_f32(const float* row, int k0, int hi) {
  Frag f;
#pragma unroll
  for (int q = 0; q < 4; ++q) {
    float4 x = *(const float4*)(row + k0 + hi * 8 + (q >> 1) * 16 + (q & 1) * 4);
    f.h[q * 4 + 0] = (__bf16)x.x;
    f.h[q * 4 + 1] = (__bf16)x.y;
    f.h[q * 4 + 2] = (__bf16)x.z;
    f.h[q * 4 + 3] = (__bf16)x.w;
  }
  return f.v;
}

// ---------------- router: top-4 + renormalized softmax ----------------
__global__ __launch_bounds__(256) void router_kernel(const float* __restrict__ logits,
                                                     int* __restrict__ topk_id,
                                                     float* __restrict__ topk_w) {
  int t = blockIdx.x * 256 + threadIdx.x;
  if (t >= T_TOK) return;
  float l[NEXP];
#pragma unroll
  for (int e = 0; e < NEXP; ++e) l[e] = logits[t * NEXP + e];
  unsigned chosen = 0u;
  int ids[TOPK];
  float vals[TOPK];
#pragma unroll
  for (int k = 0; k < TOPK; ++k) {
    float bv = -1e30f;
    int bi = 0;
#pragma unroll
    for (int e = 0; e < NEXP; ++e) {
      bool take = (((chosen >> e) & 1u) == 0u) && (l[e] > bv);
      if (take) { bv = l[e]; bi = e; }
    }
    chosen |= (1u << bi);
    ids[k] = bi;
    vals[k] = bv;
  }
  // renormalized top-k of softmax == softmax over the top-k logits
  float m = vals[0], s = 0.f, w[TOPK];
#pragma unroll
  for (int k = 0; k < TOPK; ++k) { w[k] = __expf(vals[k] - m); s += w[k]; }
  float inv = __builtin_amdgcn_rcpf(s);
#pragma unroll
  for (int k = 0; k < TOPK; ++k) {
    topk_id[t * TOPK + k] = ids[k];
    topk_w[t * TOPK + k] = w[k] * inv;
  }
}

// ------------- deterministic per-expert token-list compaction -------------
__global__ __launch_bounds__(256) void build_lists_kernel(const int* __restrict__ topk_id,
                                                          int* __restrict__ ecnt,
                                                          int* __restrict__ elist) {
  int e = blockIdx.x;
  __shared__ int pref[256];
  __shared__ int base;
  if (threadIdx.x == 0) base = 0;
  __syncthreads();
  for (int start = 0; start < T_TOK; start += 256) {
    int t = start + (int)threadIdx.x;
    int slot = -1;
#pragma unroll
    for (int k = 0; k < TOPK; ++k)
      if (topk_id[t * TOPK + k] == e) slot = k;
    int have = (slot >= 0) ? 1 : 0;
    pref[threadIdx.x] = have;
    __syncthreads();
    for (int off = 1; off < 256; off <<= 1) {
      int v = (threadIdx.x >= (unsigned)off) ? pref[threadIdx.x - off] : 0;
      __syncthreads();
      pref[threadIdx.x] += v;
      __syncthreads();
    }
    if (have) elist[e * T_TOK + base + pref[threadIdx.x] - 1] = t * TOPK + slot;
    __syncthreads();
    if (threadIdx.x == 0) base += pref[255];
    __syncthreads();
  }
  if (threadIdx.x == 0) ecnt[e] = base;
}

// ---------------- fused SwiGLU expert FFN, bf16 WMMA ----------------
__global__ __launch_bounds__(256) void moe_ffn_kernel(
    const float* __restrict__ x, const float* __restrict__ w13,
    const float* __restrict__ w2, const float* __restrict__ topk_w,
    const int* __restrict__ ecnt, const int* __restrict__ elist,
    float* __restrict__ out) {
  const int e = blockIdx.y;
  const int ne = ecnt[e];
  const int m0 = blockIdx.x * TM;
  if (m0 >= ne) return;

  extern __shared__ char smem[];
  __bf16* Xs = (__bf16*)smem;                 // TM * XS_STRIDE bf16
  __bf16* Hs = Xs + TM * XS_STRIDE;           // TM * HS_STRIDE bf16
  int* s_tok = (int*)(Hs + TM * HS_STRIDE);   // TM
  float* s_w = (float*)(s_tok + TM);          // TM

  const int tid = threadIdx.x;
  if (tid < TM) {
    int idx = m0 + tid;
    if (idx < ne) {
      int code = elist[e * T_TOK + idx];
      s_tok[tid] = code >> 2;  // token
      s_w[tid] = topk_w[code]; // combine weight
    } else {
      s_tok[tid] = 0;
      s_w[tid] = 0.f;  // padded row: contributes exactly 0
    }
  }
  __syncthreads();

  // Stage gathered X rows into LDS as bf16 (coalesced float4 reads).
  for (int i = tid; i < TM * DMODEL / 4; i += 256) {
    int lin = i * 4;
    int row = lin >> 10;           // / DMODEL
    int col = lin & (DMODEL - 1);
    float4 v = *(const float4*)(x + (size_t)s_tok[row] * DMODEL + col);
    __bf16* d = Xs + row * XS_STRIDE + col;
    d[0] = (__bf16)v.x; d[1] = (__bf16)v.y; d[2] = (__bf16)v.z; d[3] = (__bf16)v.w;
  }
  __syncthreads();

  const int wid = tid >> 5;
  const int lane = tid & 31;
  const int nlo = lane & 15;
  const int hi = lane >> 4;

  const float* w13e = w13 + (size_t)e * (2 * IDIM) * DMODEL;
  const float* w2e = w2 + (size_t)e * DMODEL * IDIM;

  const __bf16* xrow0 = Xs + (size_t)nlo * XS_STRIDE;
  const __bf16* xrow1 = Xs + (size_t)(16 + nlo) * XS_STRIDE;

  // ---- Phase 1: h[TM, I] = silu(X W13g^T) * (X W13u^T), kept in LDS ----
  for (int it = wid; it < IDIM / 16; it += NWAVE) {
    const float* grow = w13e + (size_t)(it * 16 + nlo) * DMODEL;
    const float* urow = w13e + (size_t)(IDIM + it * 16 + nlo) * DMODEL;
    v8f accG0 = zero8(), accG1 = zero8(), accU0 = zero8(), accU1 = zero8();
    for (int k0 = 0; k0 < DMODEL; k0 += 32) {
      v16bf a0 = frag_a_lds(xrow0, k0, hi);
      v16bf a1 = frag_a_lds(xrow1, k0, hi);
      v16bf bg = frag_b_f32(grow, k0, hi);
      v16bf bu = frag_b_f32(urow, k0, hi);
      accG0 = wmma_bf16(a0, bg, accG0);
      accG1 = wmma_bf16(a1, bg, accG1);
      accU0 = wmma_bf16(a0, bu, accU0);
      accU1 = wmma_bf16(a1, bu, accU1);
    }
    // C/D layout: col = lane&15, row = (lane>>4)*8 + v
#pragma unroll
    for (int v = 0; v < 8; ++v) {
      int r = hi * 8 + v;
      Hs[(size_t)r * HS_STRIDE + it * 16 + nlo] =
          (__bf16)silu_mul(accG0[v], accU0[v]);
      Hs[(size_t)(16 + r) * HS_STRIDE + it * 16 + nlo] =
          (__bf16)silu_mul(accG1[v], accU1[v]);
    }
  }
  __syncthreads();

  // ---- Phase 2: out[t, :] += c_t * (h W2^T), scatter via f32 atomics ----
  const __bf16* hrow0 = Hs + (size_t)nlo * HS_STRIDE;
  const __bf16* hrow1 = Hs + (size_t)(16 + nlo) * HS_STRIDE;
  for (int dt = wid; dt < DMODEL / 16; dt += NWAVE) {
    const float* w2row = w2e + (size_t)(dt * 16 + nlo) * IDIM;
    v8f acc0 = zero8(), acc1 = zero8();
    for (int k0 = 0; k0 < IDIM; k0 += 32) {
      v16bf a0 = frag_a_lds(hrow0, k0, hi);
      v16bf a1 = frag_a_lds(hrow1, k0, hi);
      v16bf b = frag_b_f32(w2row, k0, hi);
      acc0 = wmma_bf16(a0, b, acc0);
      acc1 = wmma_bf16(a1, b, acc1);
    }
    int dcol = dt * 16 + nlo;
#pragma unroll
    for (int v = 0; v < 8; ++v) {
      int r0 = hi * 8 + v;
      int r1 = 16 + r0;
      atomicAdd(out + (size_t)s_tok[r0] * DMODEL + dcol, s_w[r0] * acc0[v]);
      atomicAdd(out + (size_t)s_tok[r1] * DMODEL + dcol, s_w[r1] * acc1[v]);
    }
  }
}

extern "C" void kernel_launch(void* const* d_in, const int* in_sizes, int n_in,
                              void* d_out, int out_size, void* d_ws, size_t ws_size,
                              hipStream_t stream) {
  (void)in_sizes; (void)n_in; (void)out_size; (void)ws_size;
  const float* x = (const float*)d_in[0];
  const float* rl = (const float*)d_in[1];
  const float* w13 = (const float*)d_in[2];
  const float* w2 = (const float*)d_in[3];
  // d_in[4] = top_k scalar; compile-time TOPK=4.

  char* ws = (char*)d_ws;
  int* topk_id = (int*)ws;           ws += (size_t)T_TOK * TOPK * sizeof(int);
  float* topk_w = (float*)ws;        ws += (size_t)T_TOK * TOPK * sizeof(float);
  int* ecnt = (int*)ws;              ws += 256;  // NEXP ints, padded
  int* elist = (int*)ws;             // NEXP * T_TOK ints

  float* out = (float*)d_out;
  hipMemsetAsync(out, 0, (size_t)T_TOK * DMODEL * sizeof(float), stream);

  router_kernel<<<T_TOK / 256, 256, 0, stream>>>(rl, topk_id, topk_w);
  build_lists_kernel<<<NEXP, 256, 0, stream>>>(topk_id, ecnt, elist);

  size_t smem = (size_t)TM * XS_STRIDE * sizeof(__bf16) +
                (size_t)TM * HS_STRIDE * sizeof(__bf16) +
                (size_t)TM * (sizeof(int) + sizeof(float));
  // Idempotent, host-side only; safe under graph capture, no static guards.
  (void)hipFuncSetAttribute((const void*)moe_ffn_kernel,
                            hipFuncAttributeMaxDynamicSharedMemorySize,
                            (int)smem);
  moe_ffn_kernel<<<dim3(T_TOK / TM, NEXP), 256, smem, stream>>>(
      x, w13, w2, topk_w, ecnt, elist, out);
}
